// Net_13975823581217
// MI455X (gfx1250) — matrix operand order repaired and checked by reading, not compiled, not run on hardware
//
#include <hip/hip_runtime.h>
#include <hip/hip_bf16.h>
#include <math.h>

typedef __attribute__((ext_vector_type(16))) _Float16 v16h;
typedef __attribute__((ext_vector_type(8)))  float    v8f;

#define N_NODES 30000
#define N_EDGES 120000
#define N_GRAPH 512
#define H1 32
#define H2 64
#define NCLS 10
#define NN2 15000          // N/2 after pool1
#define NN3 7500           // N/4 after pool2
#define KCH 26             // 832 / 32 K-chunks
#define TILES (N_EDGES/16) // 7500, exact
#define BP_HALFS (KCH * 4 * 32 * 16) // 53248 halfs = 104 KB

__device__ __forceinline__ float eluf(float x) { return x > 0.f ? x : (expf(x) - 1.f); }

// ---------------- Layer 1: per-edge MLP + scatter-mean --------------------
__global__ void k_edge1(const float* __restrict__ x, const float* __restrict__ ea,
                        const int* __restrict__ ei,
                        const float* __restrict__ w1a, const float* __restrict__ b1a,
                        const float* __restrict__ w1b, const float* __restrict__ b1b,
                        float* __restrict__ aggr1, float* __restrict__ cnt1) {
  __shared__ float s_w1a[50], s_b1a[25], s_w1b[25 * 32], s_b1b[32];
  const int tid = threadIdx.x;
  for (int i = tid; i < 50;  i += blockDim.x) s_w1a[i] = w1a[i];
  for (int i = tid; i < 25;  i += blockDim.x) s_b1a[i] = b1a[i];
  for (int i = tid; i < 800; i += blockDim.x) s_w1b[i] = w1b[i];
  for (int i = tid; i < 32;  i += blockDim.x) s_b1b[i] = b1b[i];
  __syncthreads();
  const int e = blockIdx.x * blockDim.x + tid;
  if (e >= N_EDGES) return;
  const float a0 = ea[2 * e], a1 = ea[2 * e + 1];
  float hid[25];
#pragma unroll
  for (int k = 0; k < 25; ++k)
    hid[k] = fmaxf(a0 * s_w1a[k] + a1 * s_w1a[25 + k] + s_b1a[k], 0.f);
  const int src = ei[e], dst = ei[N_EDGES + e];
  const float xv = x[src];
#pragma unroll
  for (int o = 0; o < 32; ++o) {
    float w = s_b1b[o];
#pragma unroll
    for (int k = 0; k < 25; ++k) w += hid[k] * s_w1b[k * 32 + o];
    atomicAdd(&aggr1[dst * 32 + o], xv * w);
  }
  atomicAdd(&cnt1[dst], 1.0f);
}

__global__ void k_node1(const float* __restrict__ aggr1, const float* __restrict__ cnt1,
                        const float* __restrict__ x, const float* __restrict__ root1,
                        const float* __restrict__ bias1, float* __restrict__ h) {
  const int tid = blockIdx.x * blockDim.x + threadIdx.x;
  if (tid >= N_NODES * 32) return;
  const int v = tid >> 5, o = tid & 31;
  const float m = aggr1[tid] / fmaxf(cnt1[v], 1.0f);
  h[tid] = eluf(m + x[v] * root1[o] + bias1[o]);
}

// ---------------- Pool 1: pairwise max + mean positions -------------------
__global__ void k_pool1(const float* __restrict__ h, const float* __restrict__ pos,
                        float* __restrict__ x2, float* __restrict__ pos2) {
  const int tid = blockIdx.x * blockDim.x + threadIdx.x;
  if (tid >= NN2 * 32) return;
  const int c = tid >> 5, o = tid & 31;
  x2[tid] = fmaxf(h[(2 * c) * 32 + o], h[(2 * c + 1) * 32 + o]);
  if (o < 2)
    pos2[c * 2 + o] = 0.5f * (pos[(2 * c) * 2 + o] + pos[(2 * c + 1) * 2 + o]);
}

// ---------------- cart + global abs-max ----------------------------------
__global__ void k_cart(const int* __restrict__ ei, const float* __restrict__ pos2,
                       float* __restrict__ cart, int* __restrict__ row2,
                       int* __restrict__ col2m, unsigned int* __restrict__ maxv) {
  const int e = blockIdx.x * blockDim.x + threadIdx.x;
  if (e >= N_EDGES) return;
  const int r2 = ei[e] >> 1, c2 = ei[N_EDGES + e] >> 1;
  const bool m = (r2 != c2);
  const float cx = m ? (pos2[c2 * 2 + 0] - pos2[r2 * 2 + 0]) : 0.f;
  const float cy = m ? (pos2[c2 * 2 + 1] - pos2[r2 * 2 + 1]) : 0.f;
  cart[2 * e] = cx; cart[2 * e + 1] = cy;
  row2[e] = r2;
  col2m[e] = m ? c2 : NN2;
  atomicMax(maxv, __float_as_uint(fmaxf(fabsf(cx), fabsf(cy))));
}

__global__ void k_ea2(const float* __restrict__ cart, const unsigned int* __restrict__ maxv,
                      float* __restrict__ ea2) {
  const int e = blockIdx.x * blockDim.x + threadIdx.x;
  if (e >= N_EDGES) return;
  const float mv = __uint_as_float(*maxv);
  const float inv = (mv > 0.f) ? (0.5f / mv) : 0.f;
  ea2[2 * e]     = cart[2 * e] * inv + 0.5f;
  ea2[2 * e + 1] = cart[2 * e + 1] * inv + 0.5f;
}

// ------------- Pack Wf (832x64) into per-lane WMMA B register image ------
// Bpack[kt][nt][lane][t]: lane<16 -> (Klocal=t,     N=lane),
//                         lane>=16-> (Klocal=16+t, N=lane-16)
__global__ void k_bpack(const float* __restrict__ w2b, const float* __restrict__ b2b,
                        _Float16* __restrict__ bp) {
  const int tid = blockIdx.x * blockDim.x + threadIdx.x;
  if (tid >= BP_HALFS) return;
  const int t    = tid & 15;
  const int lane = (tid >> 4) & 31;
  const int nt   = (tid >> 9) & 3;
  const int kt   = tid >> 11;
  const int K = kt * 32 + ((lane & 16) ? 16 : 0) + t; // global K in 0..831
  const int o = nt * 16 + (lane & 15);                // output channel
  float v;
  if (K < 800) v = w2b[(K >> 5) * 2048 + (K & 31) * 64 + o]; // w2b[k, i*64+o]
  else         v = b2b[(K - 800) * 64 + o];                  // bias rows carry b2b
  bp[tid] = (_Float16)v;
}

// ------------- Layer 2 heavy kernel: fused NNConv-2 via WMMA -------------
// B (104 KB fp16) staged in LDS once per block; each wave grid-strides over
// 16-edge tiles. msg(16x64) = P(16x832) @ Wf(832x64), f32 accumulate.
__global__ void k_edge2_wmma(const float* __restrict__ ea2, const int* __restrict__ row2,
                             const int* __restrict__ col2m, const float* __restrict__ x2,
                             const float* __restrict__ w2a, const float* __restrict__ b2a,
                             const _Float16* __restrict__ bpack,
                             float* __restrict__ aggr2, float* __restrict__ cnt2) {
  __shared__ __align__(32) _Float16 sB[BP_HALFS]; // 104 KB of the 320 KB LDS

  // cooperative stage: 6656 x uint4 (16B) copies
  {
    const uint4* __restrict__ src4 = (const uint4*)bpack;
    uint4* dst4 = (uint4*)sB;
    for (int i = threadIdx.x; i < BP_HALFS / 8; i += blockDim.x) dst4[i] = src4[i];
  }
  __syncthreads();

  const int lane = threadIdx.x & 31;
  const int wave = threadIdx.x >> 5;
  const int wavesPerBlock = blockDim.x >> 5;
  const int tile0  = blockIdx.x * wavesPerBlock + wave;
  const int stride = gridDim.x * wavesPerBlock;
  const v16h* __restrict__ sbp = (const v16h*)sB;

  const int base  = (lane < 16) ? 0 : 16; // A-fragment x-slice base
  const int ncol  = lane & 15;
  const int mbase = (lane < 16) ? 0 : 8;

  // wave-uniform tile loop => EXEC stays all-ones for WMMA
  for (int tile = tile0; tile < TILES; tile += stride) {
    const int e0 = tile * 16;

    // lanes m and m+16 both own edge (e0+m): compute hidden MLP redundantly
    const int eh = e0 + (lane & 15);
    const float a0 = ea2[2 * eh], a1 = ea2[2 * eh + 1];
    float hid[25];
#pragma unroll
    for (int k = 0; k < 25; ++k)
      hid[k] = fmaxf(a0 * w2a[k] + a1 * w2a[25 + k] + b2a[k], 0.f);

    // Gather x2[row2[e]] slice matching the A-fragment lane layout:
    // lane<16: i in {0..15};  lane>=16: i in {16..31}
    const int row = row2[eh];
    const float* xr = x2 + row * 32;
    v16h xs;
#pragma unroll
    for (int t = 0; t < 16; ++t) {
      const int i = base + t;
      xs[t] = (_Float16)xr[i];
    }

    // Compiler memory barrier: keep the ds_load_b128 B-fragment reads inside
    // this loop iteration (LICM would otherwise hoist 104 fragments = 832
    // VGPRs of "loop-invariant" LDS data and spill them to scratch).
    asm volatile("" ::: "memory");

    v8f acc0 = {}, acc1 = {}, acc2 = {}, acc3 = {};
#pragma unroll
    for (int kt = 0; kt < KCH; ++kt) {
      const _Float16 hk = (kt < 25) ? (_Float16)hid[kt] : (_Float16)1.0f;
      v16h a = xs * hk;                  // A chunk = hidden[kt] * x (bias rows: *1)
      v16h b0 = sbp[(kt * 4 + 0) * 32 + lane];
      v16h b1 = sbp[(kt * 4 + 1) * 32 + lane];
      v16h b2 = sbp[(kt * 4 + 2) * 32 + lane];
      v16h b3 = sbp[(kt * 4 + 3) * 32 + lane];
      acc0 = __builtin_amdgcn_wmma_f32_16x16x32_f16(false, a, false, b0, (short)0, acc0, false, false);
      acc1 = __builtin_amdgcn_wmma_f32_16x16x32_f16(false, a, false, b1, (short)0, acc1, false, false);
      acc2 = __builtin_amdgcn_wmma_f32_16x16x32_f16(false, a, false, b2, (short)0, acc2, false, false);
      acc3 = __builtin_amdgcn_wmma_f32_16x16x32_f16(false, a, false, b3, (short)0, acc3, false, false);
    }

    // C layout: VGPR r, lane n<16 -> (M=r, N=n); lane>=16 -> (M=8+r, N=n-16)
    int dstv[8];
#pragma unroll
    for (int r = 0; r < 8; ++r) dstv[r] = col2m[e0 + mbase + r];
#pragma unroll
    for (int r = 0; r < 8; ++r) {
      atomicAdd(&aggr2[dstv[r] * 64 +  0 + ncol], acc0[r]);
      atomicAdd(&aggr2[dstv[r] * 64 + 16 + ncol], acc1[r]);
      atomicAdd(&aggr2[dstv[r] * 64 + 32 + ncol], acc2[r]);
      atomicAdd(&aggr2[dstv[r] * 64 + 48 + ncol], acc3[r]);
    }
    if (lane < 16) atomicAdd(&cnt2[col2m[e0 + lane]], 1.0f);
  }
}

__global__ void k_node2(const float* __restrict__ aggr2, const float* __restrict__ cnt2,
                        const float* __restrict__ x2, const float* __restrict__ root2,
                        const float* __restrict__ bias2, float* __restrict__ h2) {
  const int tid = blockIdx.x * blockDim.x + threadIdx.x;
  if (tid >= NN2 * 64) return;
  const int v = tid >> 6, o = tid & 63;
  float t = aggr2[v * 64 + o] / fmaxf(cnt2[v], 1.0f) + bias2[o];
  const float* xv = x2 + v * 32;
#pragma unroll
  for (int i = 0; i < 32; ++i) t += xv[i] * root2[i * 64 + o];
  h2[tid] = eluf(t);
}

__global__ void k_pool2(const float* __restrict__ h2, float* __restrict__ x3) {
  const int tid = blockIdx.x * blockDim.x + threadIdx.x;
  if (tid >= NN3 * 64) return;
  const int c = tid >> 6, o = tid & 63;
  x3[tid] = fmaxf(h2[(2 * c) * 64 + o], h2[(2 * c + 1) * 64 + o]);
}

__global__ void k_gsum(const float* __restrict__ x3, const int* __restrict__ batch,
                       float* __restrict__ gsum, float* __restrict__ gcnt) {
  const int tid = blockIdx.x * blockDim.x + threadIdx.x;
  if (tid >= NN3 * 64) return;
  const int v = tid >> 6, o = tid & 63;
  const int b = batch[4 * v]; // batch3 = batch[0::4]
  atomicAdd(&gsum[b * 64 + o], x3[tid]);
  if (o == 0) atomicAdd(&gcnt[b], 1.0f);
}

__global__ void k_fc1(const float* __restrict__ gsum, const float* __restrict__ gcnt,
                      const float* __restrict__ w, const float* __restrict__ b,
                      float* __restrict__ g1) {
  const int tid = blockIdx.x * blockDim.x + threadIdx.x;
  if (tid >= N_GRAPH * 128) return;
  const int g = tid >> 7, j = tid & 127;
  const float inv = 1.0f / fmaxf(gcnt[g], 1.0f);
  float acc = b[j];
#pragma unroll
  for (int i = 0; i < 64; ++i) acc += gsum[g * 64 + i] * inv * w[i * 128 + j];
  g1[tid] = eluf(acc);
}

__global__ void k_fc2(const float* __restrict__ g1, const float* __restrict__ w,
                      const float* __restrict__ b, float* __restrict__ out) {
  const int g = blockIdx.x * blockDim.x + threadIdx.x;
  if (g >= N_GRAPH) return;
  float z[NCLS];
#pragma unroll
  for (int c = 0; c < NCLS; ++c) z[c] = b[c];
  for (int j = 0; j < 128; ++j) {
    const float gj = g1[g * 128 + j];
#pragma unroll
    for (int c = 0; c < NCLS; ++c) z[c] += gj * w[j * NCLS + c];
  }
  float m = z[0];
#pragma unroll
  for (int c = 1; c < NCLS; ++c) m = fmaxf(m, z[c]);
  float s = 0.f;
#pragma unroll
  for (int c = 0; c < NCLS; ++c) s += expf(z[c] - m);
  const float lse = m + logf(s);
#pragma unroll
  for (int c = 0; c < NCLS; ++c) out[g * NCLS + c] = z[c] - lse;
}

extern "C" void kernel_launch(void* const* d_in, const int* in_sizes, int n_in,
                              void* d_out, int out_size, void* d_ws, size_t ws_size,
                              hipStream_t stream) {
  const float* x      = (const float*)d_in[0];
  const float* ea     = (const float*)d_in[1];
  const float* pos    = (const float*)d_in[2];
  const float* w1a    = (const float*)d_in[3];
  const float* b1a    = (const float*)d_in[4];
  const float* w1b    = (const float*)d_in[5];
  const float* b1b    = (const float*)d_in[6];
  const float* root1  = (const float*)d_in[7];
  const float* bias1  = (const float*)d_in[8];
  const float* w2a    = (const float*)d_in[9];
  const float* b2a    = (const float*)d_in[10];
  const float* w2b    = (const float*)d_in[11];
  const float* b2b    = (const float*)d_in[12];
  const float* root2  = (const float*)d_in[13];
  const float* bias2  = (const float*)d_in[14];
  const float* fc1_w  = (const float*)d_in[15];
  const float* fc1_b  = (const float*)d_in[16];
  const float* fc2_w  = (const float*)d_in[17];
  const float* fc2_b  = (const float*)d_in[18];
  const int*   ei     = (const int*)d_in[19];
  const int*   batch  = (const int*)d_in[20];
  float* out = (float*)d_out;

  // ---- workspace arena (256B aligned) ----
  char* base = (char*)d_ws;
  size_t off = 0;
  auto take = [&](size_t bytes) -> char* {
    char* p = base + off;
    off = (off + bytes + 255) & ~(size_t)255;
    return p;
  };
  float*        aggr1 = (float*)take((size_t)N_NODES * 32 * 4);
  float*        cnt1  = (float*)take((size_t)N_NODES * 4);
  float*        h     = (float*)take((size_t)N_NODES * 32 * 4);
  float*        x2    = (float*)take((size_t)NN2 * 32 * 4);
  float*        pos2  = (float*)take((size_t)NN2 * 2 * 4);
  float*        cart  = (float*)take((size_t)N_EDGES * 2 * 4);
  int*          row2  = (int*)take((size_t)N_EDGES * 4);
  int*          col2m = (int*)take((size_t)N_EDGES * 4);
  unsigned int* maxv  = (unsigned int*)take(4);
  float*        ea2   = (float*)take((size_t)N_EDGES * 2 * 4);
  _Float16*     bpack = (_Float16*)take((size_t)BP_HALFS * 2);
  float*        aggr2 = (float*)take((size_t)(NN2 + 1) * 64 * 4);
  float*        cnt2  = (float*)take((size_t)(NN2 + 1) * 4);
  float*        h2    = (float*)take((size_t)NN2 * 64 * 4);
  float*        x3    = (float*)take((size_t)NN3 * 64 * 4);
  float*        gsum  = (float*)take((size_t)N_GRAPH * 64 * 4);
  float*        gcnt  = (float*)take((size_t)N_GRAPH * 4);
  float*        g1    = (float*)take((size_t)N_GRAPH * 128 * 4);

  // ---- zero the atomic accumulation buffers (graph-capturable memsets) ----
  hipMemsetAsync(aggr1, 0, (size_t)N_NODES * 32 * 4, stream);
  hipMemsetAsync(cnt1,  0, (size_t)N_NODES * 4, stream);
  hipMemsetAsync(aggr2, 0, (size_t)(NN2 + 1) * 64 * 4, stream);
  hipMemsetAsync(cnt2,  0, (size_t)(NN2 + 1) * 4, stream);
  hipMemsetAsync(maxv,  0, 4, stream);
  hipMemsetAsync(gsum,  0, (size_t)N_GRAPH * 64 * 4, stream);
  hipMemsetAsync(gcnt,  0, (size_t)N_GRAPH * 4, stream);

  // ---- pipeline ----
  k_edge1<<<(N_EDGES + 255) / 256, 256, 0, stream>>>(x, ea, ei, w1a, b1a, w1b, b1b, aggr1, cnt1);
  k_node1<<<(N_NODES * 32 + 255) / 256, 256, 0, stream>>>(aggr1, cnt1, x, root1, bias1, h);
  k_pool1<<<(NN2 * 32 + 255) / 256, 256, 0, stream>>>(h, pos, x2, pos2);
  k_cart<<<(N_EDGES + 255) / 256, 256, 0, stream>>>(ei, pos2, cart, row2, col2m, maxv);
  k_ea2<<<(N_EDGES + 255) / 256, 256, 0, stream>>>(cart, maxv, ea2);
  k_bpack<<<(BP_HALFS + 255) / 256, 256, 0, stream>>>(w2b, b2b, bpack);
  // 8 waves/block, B staged in 104 KB LDS, ~32 tiles amortize each staging
  k_edge2_wmma<<<235, 256, 0, stream>>>(ea2, row2, col2m, x2, w2a, b2a, bpack, aggr2, cnt2);
  k_node2<<<(NN2 * 64 + 255) / 256, 256, 0, stream>>>(aggr2, cnt2, x2, root2, bias2, h2);
  k_pool2<<<(NN3 * 64 + 255) / 256, 256, 0, stream>>>(h2, x3);
  k_gsum<<<(NN3 * 64 + 255) / 256, 256, 0, stream>>>(x3, batch, gsum, gcnt);
  k_fc1<<<(N_GRAPH * 128 + 255) / 256, 256, 0, stream>>>(gsum, gcnt, fc1_w, fc1_b, g1);
  k_fc2<<<(N_GRAPH + 255) / 256, 256, 0, stream>>>(g1, fc2_w, fc2_b, out);
}
